// MultiHeadLatentAttention_82291573391984
// MI455X (gfx1250) — compile-verified, matrix operand order checked by reading, not run on hardware
//
#include <hip/hip_runtime.h>

// ---------------------------------------------------------------------------
// MLA (DeepSeek-style multi-head latent attention) prefill for gfx1250.
// All GEMMs + attention matmuls run on v_wmma_f32_16x16x32_f16 (wave32).
// B=2, S=2048, DIM=2048, H=16, q_rank=1536, kv_rank=512, d_nope=128,
// d_rope=64, d_v=128.  M = B*S = 4096 tokens for every projection.
// ---------------------------------------------------------------------------

typedef _Float16 half_t;
typedef __attribute__((ext_vector_type(16))) _Float16 v16h;
typedef __attribute__((ext_vector_type(8)))  float    v8f;

union FragU {
  v16h  v;
  uint4 q[2];
  half_t h[16];
};

__device__ __forceinline__ v8f wmma16(v16h a, v16h b, v8f c) {
  // 8 args: (neg_a, A, neg_b, B, c_mod, C, reuse_a, reuse_b)
  return __builtin_amdgcn_wmma_f32_16x16x32_f16(false, a, false, b,
                                                (short)0, c, false, false);
}

// Load a 16x32 f16 fragment from a row-major [rows, stride] array.
// ISA 16-bit A layout: lane = m + 16*((k>>3)&1); per lane the 16 halves are
// two contiguous 8-half (16B) chunks at k = hi*8.. and k = 16+hi*8..
// B fragments of A*B^T use the same pattern with n in place of m.
__device__ __forceinline__ v16h load_frag(const half_t* base, int stride, int lane) {
  const int row = lane & 15;
  const int hi  = lane >> 4;
  const half_t* p = base + (size_t)row * stride + hi * 8;
  FragU u;
  u.q[0] = *(const uint4*)(p);
  u.q[1] = *(const uint4*)(p + 16);
  return u.v;
}

// ---------------------------------------------------------------------------
// Generic GEMM: C[M,N] = A[M,K] * B[N,K]^T, f16 inputs, f32 accumulate.
// Block tile 256(M) x 64(N), 256 threads = 8 waves, each wave owns a
// 32x64 sub-tile (2 m-frags x 4 n-frags = 8 WMMA per 32-wide K-step).
// ---------------------------------------------------------------------------
template <bool OUT_F16>
__global__ __launch_bounds__(256) void gemm_nt_wmma(
    const half_t* __restrict__ A, const half_t* __restrict__ B,
    void* __restrict__ C, int M, int N, int K)
{
  __shared__ __align__(16) half_t As[256 * 32];
  __shared__ __align__(16) half_t Bs[64 * 32];

  const int tid  = threadIdx.x;
  const int lane = tid & 31;
  const int wave = tid >> 5;             // 0..7 -> 32 M-rows each
  const int mBase = blockIdx.y * 256;
  const int nBase = blockIdx.x * 64;

  v8f z = {};
  v8f acc[2][4];
#pragma unroll
  for (int mf = 0; mf < 2; ++mf)
#pragma unroll
    for (int nf = 0; nf < 4; ++nf) acc[mf][nf] = z;

  const int brow = tid >> 2, bkc = (tid & 3) * 8;
  const half_t* aSrc = A + (size_t)(mBase + tid) * K;          // 1 row/thread
  const half_t* bSrc = B + (size_t)(nBase + brow) * K + bkc;

  for (int kk = 0; kk < K; kk += 32) {
    *(uint4*)&As[tid * 32 +  0] = *(const uint4*)(aSrc + kk);
    *(uint4*)&As[tid * 32 +  8] = *(const uint4*)(aSrc + kk + 8);
    *(uint4*)&As[tid * 32 + 16] = *(const uint4*)(aSrc + kk + 16);
    *(uint4*)&As[tid * 32 + 24] = *(const uint4*)(aSrc + kk + 24);
    *(uint4*)&Bs[brow * 32 + bkc] = *(const uint4*)(bSrc + kk);
    if (kk + 32 < K) {
      __builtin_prefetch(aSrc + kk + 32, 0, 1);   // global_prefetch_b8
      __builtin_prefetch(bSrc + kk + 32, 0, 1);
    }
    __syncthreads();

    v16h a0 = load_frag(&As[(wave * 32 +  0) * 32], 32, lane);
    v16h a1 = load_frag(&As[(wave * 32 + 16) * 32], 32, lane);
    v16h bf[4];
#pragma unroll
    for (int nf = 0; nf < 4; ++nf)
      bf[nf] = load_frag(&Bs[(nf * 16) * 32], 32, lane);
#pragma unroll
    for (int nf = 0; nf < 4; ++nf) {
      acc[0][nf] = wmma16(a0, bf[nf], acc[0][nf]);
      acc[1][nf] = wmma16(a1, bf[nf], acc[1][nf]);
    }
    __syncthreads();
  }

  // D tile layout: element (m,n) at vgpr r, lane L with m = r + 8*(L>>4),
  // n = L&15.
  const int hi = lane >> 4, n = lane & 15;
#pragma unroll
  for (int mf = 0; mf < 2; ++mf) {
#pragma unroll
    for (int nf = 0; nf < 4; ++nf) {
      const int col  = nBase + nf * 16 + n;
      const int row0 = mBase + wave * 32 + mf * 16 + hi * 8;
#pragma unroll
      for (int r = 0; r < 8; ++r) {
        const size_t idx = (size_t)(row0 + r) * N + col;
        const float val = acc[mf][nf][r];
        if (OUT_F16) ((half_t*)C)[idx] = (half_t)val;
        else         ((float*)C)[idx]  = val;
      }
    }
  }
}

// ---------------------------------------------------------------------------
__global__ void cvt_f32_to_f16(const float* __restrict__ in,
                               half_t* __restrict__ out, size_t n)
{
  size_t i = (size_t)blockIdx.x * blockDim.x + threadIdx.x;
  const size_t stride = (size_t)gridDim.x * blockDim.x;
  for (; i < n; i += stride) out[i] = (half_t)in[i];
}

// One row per block; f32 in, f16 normalized out.
__global__ __launch_bounds__(256) void rmsnorm_to_f16(
    const float* __restrict__ in, const float* __restrict__ w,
    half_t* __restrict__ out, int C)
{
  __shared__ float red[256];
  const int row = blockIdx.x, tid = threadIdx.x;
  const float* x = in + (size_t)row * C;
  float ss = 0.f;
  for (int i = tid; i < C; i += 256) { float v = x[i]; ss += v * v; }
  red[tid] = ss;
  __syncthreads();
  for (int s = 128; s > 0; s >>= 1) {
    if (tid < s) red[tid] += red[tid + s];
    __syncthreads();
  }
  const float scale = rsqrtf(red[0] / (float)C + 1e-6f);
  half_t* o = out + (size_t)row * C;
  for (int i = tid; i < C; i += 256) o[i] = (half_t)(x[i] * scale * w[i]);
}

// ---------------------------------------------------------------------------
// Assemble per-head Q[b,h,s,192], K[b,h,s,192] (f16) with interleaved-pair
// RoPE on the 64-d rope slices (shared rope key broadcast across heads),
// and V TRANSPOSED as Vt[b,h,d=128,s=2048] so attention's P*V B-fragments
// are contiguous 16B loads.  One block per (s,h,b), 128 threads.
// ---------------------------------------------------------------------------
__global__ __launch_bounds__(128) void pack_qkv(
    const half_t* __restrict__ qnope, const half_t* __restrict__ qpe,
    const half_t* __restrict__ kvup,  const float* __restrict__ krope,
    half_t* __restrict__ Q, half_t* __restrict__ Kb, half_t* __restrict__ Vt)
{
  const int s = blockIdx.x, h = blockIdx.y, b = blockIdx.z;
  const int tid = threadIdx.x;
  const size_t token = (size_t)b * 2048 + s;
  const size_t bh = (size_t)b * 16 + h;
  const size_t qk = (bh * 2048 + s) * 192;

  Q [qk + tid] = qnope[token * 2048 + (size_t)h * 128 + tid];
  Kb[qk + tid] = kvup [token * 4096 + (size_t)h * 256 + tid];
  // transposed V: Vt[bh][d][s]
  Vt[(bh * 128 + tid) * 2048 + s] = kvup[token * 4096 + (size_t)h * 256 + 128 + tid];

  if (tid < 32) {
    const int i = tid;                       // pair index, d = 64
    // inv_freq = 10000^(-2i/64) = exp(-2i/64 * ln 10000)
    const float inv_freq = __expf(-(float)(2 * i) * (9.210340371976184f / 64.0f));
    const float ang = (float)s * inv_freq;
    const float c = cosf(ang), sn = sinf(ang);

    const float x1 = (float)qpe[token * 1024 + (size_t)h * 64 + 2 * i];
    const float x2 = (float)qpe[token * 1024 + (size_t)h * 64 + 2 * i + 1];
    Q[qk + 128 + 2 * i]     = (half_t)(x1 * c - x2 * sn);
    Q[qk + 128 + 2 * i + 1] = (half_t)(x1 * sn + x2 * c);

    const float y1 = krope[token * 64 + 2 * i];
    const float y2 = krope[token * 64 + 2 * i + 1];
    Kb[qk + 128 + 2 * i]     = (half_t)(y1 * c - y2 * sn);
    Kb[qk + 128 + 2 * i + 1] = (half_t)(y1 * sn + y2 * c);
  }
}

// ---------------------------------------------------------------------------
// Causal flash attention.  One block = 64 queries of one (b,h); 4 fully
// independent waves (no barriers), each owning a 16-query strip:
//   QK^T: 12 wmma / 32-key step, K fragments straight from global memory.
//   f32 online softmax with 16-lane shfl reductions.
//   P through per-wave LDS to re-shape D-tile -> A-fragment.
//   P*V: 8 wmma / step, V fragments straight from the transposed Vt buffer.
// ---------------------------------------------------------------------------
__global__ __launch_bounds__(128) void mla_flash_attn(
    const half_t* __restrict__ Q, const half_t* __restrict__ Kb,
    const half_t* __restrict__ Vt, half_t* __restrict__ Out)
{
  __shared__ __align__(16) half_t Ps[4][16 * 32];

  const int tid = threadIdx.x, lane = tid & 31, wave = tid >> 5;
  const int b = blockIdx.z, h = blockIdx.y;
  const int qb = blockIdx.x * 64;
  const size_t bh = (size_t)b * 16 + h;
  const half_t* Qp  = Q  + (bh * 2048 + (size_t)qb + wave * 16) * 192;
  const half_t* Kp  = Kb + bh * 2048 * 192;
  const half_t* Vtp = Vt + bh * 128 * 2048;
  const float SCALE = 0.07216878364870323f;  // 1/sqrt(192)

  v16h qf[6];
#pragma unroll
  for (int f = 0; f < 6; ++f) qf[f] = load_frag(Qp + f * 32, 192, lane);

  v8f z = {};
  v8f o[8];
#pragma unroll
  for (int c = 0; c < 8; ++c) o[c] = z;
  float mrow[8], lrow[8];
#pragma unroll
  for (int r = 0; r < 8; ++r) { mrow[r] = -1e30f; lrow[r] = 0.f; }

  const int hi = lane >> 4, n = lane & 15;
  const int nk = (qb + 64) / 32;             // causal bound for this q tile

  for (int step = 0; step < nk; ++step) {
    const int kbase = step * 32;

    // S = Q * K^T for two 16-key subtiles; K fragments read contiguously
    // from the row-major K buffer (A*B^T trick).
    v8f s0 = z, s1 = z;
#pragma unroll
    for (int f = 0; f < 6; ++f) {
      v16h k0 = load_frag(Kp + (size_t)(kbase)      * 192 + f * 32, 192, lane);
      v16h k1 = load_frag(Kp + (size_t)(kbase + 16) * 192 + f * 32, 192, lane);
      s0 = wmma16(qf[f], k0, s0);
      s1 = wmma16(qf[f], k1, s1);
    }

    float alpha[8];
#pragma unroll
    for (int r = 0; r < 8; ++r) {
      const int qrow = qb + wave * 16 + hi * 8 + r;
      float v0 = s0[r] * SCALE, v1 = s1[r] * SCALE;
      if (kbase + n      > qrow) v0 = -1e30f;     // causal mask
      if (kbase + 16 + n > qrow) v1 = -1e30f;
      float mx = fmaxf(v0, v1);
#pragma unroll
      for (int off = 1; off < 16; off <<= 1)      // 16-lane row-max
        mx = fmaxf(mx, __shfl_xor(mx, off, 32));
      const float mnew = fmaxf(mrow[r], mx);
      const float a  = __expf(mrow[r] - mnew);
      const float p0 = __expf(v0 - mnew);
      const float p1 = __expf(v1 - mnew);
      float ls = p0 + p1;
#pragma unroll
      for (int off = 1; off < 16; off <<= 1)      // 16-lane row-sum
        ls += __shfl_xor(ls, off, 32);
      lrow[r] = lrow[r] * a + ls;
      mrow[r] = mnew;
      alpha[r] = a;
      s0[r] = p0; s1[r] = p1;
    }

    // P: D-tile layout -> row-major LDS -> A-fragment.  Same-wave LDS ops
    // are in order, so no workgroup barrier is needed for Ps[wave].
    half_t* pw = &Ps[wave][0];
#pragma unroll
    for (int r = 0; r < 8; ++r) {
      pw[(hi * 8 + r) * 32 + n]      = (half_t)s0[r];
      pw[(hi * 8 + r) * 32 + n + 16] = (half_t)s1[r];
    }
    v16h pf = load_frag(pw, 32, lane);

#pragma unroll
    for (int c = 0; c < 8; ++c)
#pragma unroll
      for (int r = 0; r < 8; ++r) o[c][r] *= alpha[r];

    // O += P(16x32) * V(32x128): frag[k][n] = Vt[c*16+n][kbase+k] -> two
    // contiguous 16B global loads per lane from the transposed V buffer.
#pragma unroll
    for (int c = 0; c < 8; ++c) {
      v16h vf = load_frag(Vtp + (size_t)(c * 16) * 2048 + kbase, 2048, lane);
      o[c] = wmma16(pf, vf, o[c]);
    }
  }

  // Write attn_out[token, h*128 + d] (f16), dividing by the softmax sum.
  const size_t token0 = (size_t)b * 2048 + qb + wave * 16;
#pragma unroll
  for (int r = 0; r < 8; ++r) {
    const float inv_l = 1.0f / lrow[r];
    const size_t orow = (token0 + hi * 8 + r) * 2048 + (size_t)h * 128;
#pragma unroll
    for (int c = 0; c < 8; ++c)
      Out[orow + c * 16 + n] = (half_t)(o[c][r] * inv_l);
  }
}

// ---------------------------------------------------------------------------
extern "C" void kernel_launch(void* const* d_in, const int* in_sizes, int n_in,
                              void* d_out, int out_size, void* d_ws, size_t ws_size,
                              hipStream_t stream)
{
  (void)in_sizes; (void)n_in; (void)out_size; (void)ws_size;
  const float* x         = (const float*)d_in[0];
  const float* wq_down   = (const float*)d_in[1];
  const float* q_norm_w  = (const float*)d_in[2];
  const float* wq_up     = (const float*)d_in[3];
  const float* wq_rope   = (const float*)d_in[4];
  const float* wkv_down  = (const float*)d_in[5];
  const float* kv_norm_w = (const float*)d_in[6];
  const float* wkv_up    = (const float*)d_in[7];
  const float* wk_rope   = (const float*)d_in[8];
  const float* wo        = (const float*)d_in[9];
  float* out = (float*)d_out;

  const int M = 4096;  // B*S tokens

  char* wsp = (char*)d_ws;
  auto alloc = [&](size_t bytes) -> char* {
    char* p = wsp;
    wsp += (bytes + 255) & ~(size_t)255;
    return p;
  };

  half_t* x_h       = (half_t*)alloc((size_t)M * 2048 * 2);
  half_t* wqd_h     = (half_t*)alloc((size_t)1536 * 2048 * 2);
  half_t* wqu_h     = (half_t*)alloc((size_t)2048 * 1536 * 2);
  half_t* wqr_h     = (half_t*)alloc((size_t)1024 * 1536 * 2);
  half_t* wkvd_h    = (half_t*)alloc((size_t)512 * 2048 * 2);
  half_t* wkvu_h    = (half_t*)alloc((size_t)4096 * 512 * 2);
  half_t* wkr_h     = (half_t*)alloc((size_t)64 * 2048 * 2);
  half_t* wo_h      = (half_t*)alloc((size_t)2048 * 2048 * 2);
  float*  qc_pre    = (float*) alloc((size_t)M * 1536 * 4);
  float*  kvc_pre   = (float*) alloc((size_t)M * 512 * 4);
  float*  krope_pre = (float*) alloc((size_t)M * 64 * 4);
  half_t* qc_h      = (half_t*)alloc((size_t)M * 1536 * 2);
  half_t* kvc_h     = (half_t*)alloc((size_t)M * 512 * 2);
  half_t* qnope_h   = (half_t*)alloc((size_t)M * 2048 * 2);
  half_t* qpe_h     = (half_t*)alloc((size_t)M * 1024 * 2);
  half_t* kvup_h    = (half_t*)alloc((size_t)M * 4096 * 2);
  half_t* Qb        = (half_t*)alloc((size_t)2 * 16 * 2048 * 192 * 2);
  half_t* Kbuf      = (half_t*)alloc((size_t)2 * 16 * 2048 * 192 * 2);
  half_t* Vtbuf     = (half_t*)alloc((size_t)2 * 16 * 128 * 2048 * 2);
  half_t* attn_h    = (half_t*)alloc((size_t)M * 2048 * 2);

  auto cvt = [&](const float* src, half_t* dst, size_t nelem) {
    int blocks = (int)((nelem + 1023) / 1024);
    if (blocks > 4096) blocks = 4096;
    if (blocks < 1) blocks = 1;
    cvt_f32_to_f16<<<blocks, 256, 0, stream>>>(src, dst, nelem);
  };
  cvt(x,        x_h,    (size_t)M * 2048);
  cvt(wq_down,  wqd_h,  (size_t)1536 * 2048);
  cvt(wq_up,    wqu_h,  (size_t)2048 * 1536);
  cvt(wq_rope,  wqr_h,  (size_t)1024 * 1536);
  cvt(wkv_down, wkvd_h, (size_t)512 * 2048);
  cvt(wkv_up,   wkvu_h, (size_t)4096 * 512);
  cvt(wk_rope,  wkr_h,  (size_t)64 * 2048);
  cvt(wo,       wo_h,   (size_t)2048 * 2048);

  // Down projections + shared rope key (f32 out, pre-norm).
  gemm_nt_wmma<false><<<dim3(1536 / 64, M / 256), 256, 0, stream>>>(x_h, wqd_h,  qc_pre,    M, 1536, 2048);
  gemm_nt_wmma<false><<<dim3(512 / 64,  M / 256), 256, 0, stream>>>(x_h, wkvd_h, kvc_pre,   M, 512,  2048);
  gemm_nt_wmma<false><<<dim3(64 / 64,   M / 256), 256, 0, stream>>>(x_h, wkr_h,  krope_pre, M, 64,   2048);

  rmsnorm_to_f16<<<M, 256, 0, stream>>>(qc_pre,  q_norm_w,  qc_h,  1536);
  rmsnorm_to_f16<<<M, 256, 0, stream>>>(kvc_pre, kv_norm_w, kvc_h, 512);

  // Up projections (f16 out, consumed by pack/attention).
  gemm_nt_wmma<true><<<dim3(2048 / 64, M / 256), 256, 0, stream>>>(qc_h,  wqu_h,  qnope_h, M, 2048, 1536);
  gemm_nt_wmma<true><<<dim3(1024 / 64, M / 256), 256, 0, stream>>>(qc_h,  wqr_h,  qpe_h,   M, 1024, 1536);
  gemm_nt_wmma<true><<<dim3(4096 / 64, M / 256), 256, 0, stream>>>(kvc_h, wkvu_h, kvup_h,  M, 4096, 512);

  pack_qkv<<<dim3(2048, 16, 2), 128, 0, stream>>>(qnope_h, qpe_h, kvup_h, krope_pre, Qb, Kbuf, Vtbuf);

  mla_flash_attn<<<dim3(2048 / 64, 16, 2), 128, 0, stream>>>(Qb, Kbuf, Vtbuf, attn_h);

  // Output projection -> f32 d_out.
  gemm_nt_wmma<false><<<dim3(2048 / 64, M / 256), 256, 0, stream>>>(attn_h, wo_h, out, M, 2048, 2048);
}
